// Model_73821897883893
// MI455X (gfx1250) — matrix operand order repaired and checked by reading, not compiled, not run on hardware
//
#include <hip/hip_runtime.h>
#include <hip/hip_bf16.h>

// ---------------------------------------------------------------------------
// RNTN forward on MI455X (gfx1250, wave32).
// Hoist the h-independent contraction Y[b,s,(i,j)] = sum_e X[b,s,e]*A[i,e,j]
// (~98.6 GFLOPs, >95% of work) into a bf16 V_WMMA_F32_16X16X32_BF16 GEMM,
// fold x@W into the same GEMM as 25 extra N-columns, then run the cheap
// sequential tanh recurrence (parallel over batch x D) and the sigmoid head.
// GEMM tiling: 1 M-tile x 4 N-tiles per wave, N padded to 44 tiles so the
// K x N loop nest is branch-free (EXEC all-1s for every WMMA, ISA 7.12).
// All 4 B fragments of a K-step are preloaded before the WMMA quartet so
// loads stay in flight under the WMMAs (partial s_wait_loadcnt, not drains).
// ---------------------------------------------------------------------------

typedef __bf16 v16bf __attribute__((ext_vector_type(16)));
typedef float  v8f   __attribute__((ext_vector_type(8)));

#define S_LEN  128
#define E_DIM  301
#define E_PAD  320          // K padded to multiple of 32
#define D_DIM  25
#define DD     625          // D*D
#define N_TOT  650          // 625 bilinear cols + 25 W cols
#define N_PAD  704          // padded to 44 tiles = 11 groups of exactly 4
#define NT_GROUPS 11
#define NCLS   5

struct U8 { unsigned int u[8]; };

// --- one-time prep: Bt[n][e] (column-major of the fused [E_PAD, N_PAD] B) ---
__global__ void prep_B(const float* __restrict__ A, const float* __restrict__ W,
                       __bf16* __restrict__ Bt) {
  int idx = blockIdx.x * blockDim.x + threadIdx.x;
  if (idx >= N_PAD * E_PAD) return;
  int n = idx / E_PAD, e = idx % E_PAD;
  float v = 0.f;
  if (e < E_DIM) {
    if (n < DD) {                      // bilinear block: n = i*25 + j
      int i = n / D_DIM, j = n % D_DIM;
      v = A[(i * E_DIM + e) * D_DIM + j];        // A[i,e,j], A is [D,E,D]
    } else if (n < N_TOT) {
      v = W[e * D_DIM + (n - DD)];               // W[e,d],  W is [E,D]
    }
  }
  Bt[(size_t)n * E_PAD + e] = (__bf16)v;
}

// --- per-chunk embedding gather -> padded bf16 activation matrix -----------
__global__ void gather_x(const int* __restrict__ words,
                         const float* __restrict__ emb,
                         __bf16* __restrict__ Xbf,
                         int chunk_base, int rows) {
  int idx = blockIdx.x * blockDim.x + threadIdx.x;
  if (idx >= rows * E_PAD) return;
  int row = idx / E_PAD, e = idx % E_PAD;
  float v = 0.f;
  if (e < E_DIM) {
    int b = chunk_base + row / S_LEN;
    int s = row % S_LEN;
    int w = words[b * S_LEN + s];
    v = emb[(size_t)w * E_DIM + e];    // 60MB table -> lives in 192MB L2
  }
  Xbf[(size_t)row * E_PAD + e] = (__bf16)v;
}

// --- the big GEMM: 16x64 output per wave, branch-free 10 x 4 WMMA stream ---
// Grid is sized exactly (mtiles * NT_GROUPS waves), so no guard is needed.
__global__ void __launch_bounds__(256)
gemm_wmma(const __bf16* __restrict__ Xbf, const __bf16* __restrict__ Bt,
          float* __restrict__ Y) {
  int wave = (int)(blockIdx.x * blockDim.x + threadIdx.x) >> 5;
  int lane = threadIdx.x & 31;
  int mt  = wave / NT_GROUPS;
  int nt0 = (wave % NT_GROUPS) * 4;
  int lh  = lane >> 4;                           // lane half (K-group select)
  int l16 = lane & 15;

  const __bf16* arow  = Xbf + (size_t)(mt * 16 + l16) * E_PAD;    // M = l16
  const __bf16* bcol0 = Bt  + (size_t)(nt0 * 16 + l16) * E_PAD;   // N = l16

  v8f c[4];
#pragma unroll
  for (int t = 0; t < 4; ++t) c[t] = (v8f){0.f,0.f,0.f,0.f,0.f,0.f,0.f,0.f};

#pragma unroll
  for (int kb = 0; kb < E_PAD; kb += 32) {
    // A frag: VGPR p -> K = lh*16 + 2p(+1), matching the B fragment order.
    // Pairs are memory-contiguous.
    U8 ua;
    const __bf16* ap = arow + kb + lh * 16;
    __builtin_prefetch(ap + 32, 0, 1);           // global_prefetch_b8 next K slab
#pragma unroll
    for (int p = 0; p < 4; ++p) {
      ua.u[p]     = *(const unsigned int*)(ap + 2 * p);
      ua.u[p + 4] = *(const unsigned int*)(ap + 8 + 2 * p);
    }

    // Preload ALL 4 B fragments first so 10 b128 loads are in flight before
    // the first WMMA; the quartet then drains them with partial waits.
    // B frag: half-slot h (=2p+lo) <-> K = h + lh*16; Bt column-major so the
    // 16 K-values per lane are one contiguous 32B run.
    U8 ub[4];
#pragma unroll
    for (int t = 0; t < 4; ++t) {
      const __bf16* bp = bcol0 + (size_t)t * 16 * E_PAD + kb + lh * 16;
#pragma unroll
      for (int p = 0; p < 8; ++p)
        ub[t].u[p] = *(const unsigned int*)(bp + 2 * p);
    }

    v16bf a = __builtin_bit_cast(v16bf, ua);
#pragma unroll
    for (int t = 0; t < 4; ++t) {
      v16bf b = __builtin_bit_cast(v16bf, ub[t]);
      c[t] = __builtin_amdgcn_wmma_f32_16x16x32_bf16(
          /*neg_a=*/false, a, /*neg_b=*/false, b,
          /*c_mod=*/(short)0, c[t], /*reuse_a=*/false, /*reuse_b=*/false);
    }
  }

  // C/D layout: VGPR r -> M = r + lh*8, N = l16
#pragma unroll
  for (int t = 0; t < 4; ++t) {
    float* yout = Y + (size_t)(mt * 16 + lh * 8) * N_PAD + (nt0 + t) * 16 + l16;
#pragma unroll
    for (int r = 0; r < 8; ++r)
      yout[(size_t)r * N_PAD] = c[t][r];
  }
}

// --- sequential recurrence + sigmoid head: 1 wave per batch element --------
__global__ void __launch_bounds__(256)
recur(const float* __restrict__ Y, const float* __restrict__ V,
      const float* __restrict__ bvec, const float* __restrict__ outW,
      const float* __restrict__ outb, float* __restrict__ out, int chunk_base) {
  __shared__ float Vs[DD];
  __shared__ float hs[8][D_DIM + 1];
  int tid = threadIdx.x;
  for (int t = tid; t < DD; t += 256) Vs[t] = V[t];
  int g = tid >> 5, lane = tid & 31;
  int b_local = blockIdx.x * 8 + g;
  if (lane < D_DIM) hs[g][lane] = (lane == D_DIM - 1) ? 1.f : 0.f;  // h0
  float bi = (lane < D_DIM) ? bvec[lane] : 0.f;
  __syncthreads();

  for (int s = 0; s < S_LEN; ++s) {
    const float* yrow = Y + (size_t)(b_local * S_LEN + s) * N_PAD;
    float hn = 0.f;
    if (lane < D_DIM) {
      float acc = yrow[DD + lane] + bi;           // x@W column + bias
#pragma unroll
      for (int j = 0; j < D_DIM; ++j) {
        float hj = hs[g][j];
        acc += yrow[lane * D_DIM + j] * hj;       // bilinear: Y[i,j] * h[j]
        acc += hj * Vs[j * D_DIM + lane];         // h @ V
      }
      hn = tanhf(acc);
    }
    __syncthreads();
    if (lane < D_DIM) hs[g][lane] = hn;
    __syncthreads();
  }

  if (lane < NCLS) {
    float lg = outb[lane];
#pragma unroll
    for (int j = 0; j < D_DIM; ++j) lg += hs[g][j] * outW[lane * D_DIM + j];
    out[(size_t)(chunk_base + b_local) * NCLS + lane] = 1.f / (1.f + expf(-lg));
  }
}

extern "C" void kernel_launch(void* const* d_in, const int* in_sizes, int n_in,
                              void* d_out, int out_size, void* d_ws, size_t ws_size,
                              hipStream_t stream) {
  const int*   words = (const int*)d_in[0];
  const float* emb   = (const float*)d_in[2];
  const float* A     = (const float*)d_in[3];
  const float* W     = (const float*)d_in[4];
  const float* V     = (const float*)d_in[5];
  const float* bvec  = (const float*)d_in[6];
  const float* outW  = (const float*)d_in[7];
  const float* outb  = (const float*)d_in[8];
  float*       out   = (float*)d_out;

  int B = in_sizes[0] / S_LEN;                    // 2048

  size_t btBytes = (size_t)N_PAD * E_PAD * sizeof(__bf16);   // 0.45 MB, 16B-aligned
  // Pick largest batch chunk whose X(bf16)+Y(f32) staging fits the workspace.
  int Bc = 128;
  while (Bc > 8) {
    size_t need = btBytes + (size_t)Bc * S_LEN * E_PAD * 2
                          + (size_t)Bc * S_LEN * N_PAD * 4;
    if (need <= ws_size && (B % Bc) == 0) break;
    Bc >>= 1;
  }

  char*   wsp = (char*)d_ws;
  __bf16* Bt  = (__bf16*)wsp;
  __bf16* Xbf = (__bf16*)(wsp + btBytes);
  float*  Y   = (float*)(wsp + btBytes + (size_t)Bc * S_LEN * E_PAD * 2);

  {
    int tot = N_PAD * E_PAD;
    prep_B<<<(tot + 255) / 256, 256, 0, stream>>>(A, W, Bt);
  }

  int rows   = Bc * S_LEN;
  int mtiles = rows / 16;                         // multiple of 64 (Bc >= 8)
  int nChunks = B / Bc;
  for (int c = 0; c < nChunks; ++c) {
    int base = c * Bc;
    int totg = rows * E_PAD;
    gather_x<<<(totg + 255) / 256, 256, 0, stream>>>(words, emb, Xbf, base, rows);
    int threads = mtiles * NT_GROUPS * 32;        // exact: multiple of 256
    gemm_wmma<<<threads / 256, 256, 0, stream>>>(Xbf, Bt, Y);
    recur<<<Bc / 8, 256, 0, stream>>>(Y, V, bvec, outW, outb, out, base);
  }
}
